// SequenceGRU_11106785427625
// MI455X (gfx1250) — matrix operand order repaired and checked by reading, not compiled, not run on hardware
//
#include <hip/hip_runtime.h>

// ---------------------------------------------------------------------------
// 2-layer LSTM scan for MI455X (gfx1250), fp16 WMMA + fp32 accumulate.
//   B=64, S=1024 (1023 scanned steps), I=300 (pad 320), H=1024, O=100 (pad 112)
// Phase 1: xg1[t,b,g] = xs @ W_ih1^T + (b_ih1+b_hh1)      (one big WMMA GEMM)
// Phase 2: ONE fused kernel per pipeline tick t:
//            blocks   0..63  : layer-1 step t        (if t < TT)
//            blocks  64..127 : layer-2 step t-1      (if 1 <= t <= TT)
//            blocks 128..134 : out-proj step t-2     (if t >= 2)
//          All intra-launch shared-buffer accesses are read-read; the h1/h2
//          double buffers carry producer->consumer deps across launches.
//          Weights stay hot in the 192MB L2; hidden state staged via LDS.
// ---------------------------------------------------------------------------

typedef __attribute__((ext_vector_type(16))) _Float16     v16h;
typedef __attribute__((ext_vector_type(8)))  float        v8f;
typedef __attribute__((ext_vector_type(4)))  unsigned int v4u;

#define BB    64          // batch
#define HH    1024        // hidden
#define GG    4096        // 4*H gates
#define II    300
#define IPAD  320
#define SS    1024
#define TT    1023        // scanned steps
#define OO    100
#define OPAD  112
#define ROWS  (TT * BB)   // 65472 rows of the precompute GEMM

union Frag16 { v16h h; v4u q[2]; };

// A-matrix fragment (16x32 f16). src row-major, ld in halfs.
// lane L: row = rowBase + L%16 ; K pairs at (v/4)*16 + (L/16)*8 + (v%4)*2
__device__ __forceinline__ v16h load_a(const _Float16* __restrict__ src,
                                       int rowBase, int ld, int kBase,
                                       int l16, int lh) {
  Frag16 f;
  const _Float16* p = src + (size_t)(rowBase + l16) * ld + kBase + lh * 8;
  f.q[0] = *(const v4u*)(p);
  f.q[1] = *(const v4u*)(p + 16);
  return f.h;
}

// B-matrix fragment (32x16 f16) with B[k][n] = W[colBase+n][k], W row-major.
// lane L: col = colBase + L%16 ; K pairs at (v/4)*16 + (L/16)*8 + (v%4)*2
__device__ __forceinline__ v16h load_b(const _Float16* __restrict__ W,
                                       int colBase, int ld, int kBase,
                                       int l16, int lh) {
  Frag16 f;
  const _Float16* p = W + (size_t)(colBase + l16) * ld + kBase + lh * 8;
  f.q[0] = *(const v4u*)(p);
  f.q[1] = *(const v4u*)(p + 16);
  return f.h;
}

__device__ __forceinline__ v8f wmma16(v16h a, v16h b, v8f c) {
  return __builtin_amdgcn_wmma_f32_16x16x32_f16(false, a, false, b,
                                                (short)0, c, false, false);
}

__device__ __forceinline__ float fsigmoid(float x) {
  return 1.0f / (1.0f + __expf(-x));
}
__device__ __forceinline__ float ftanh(float x) {
  x = fminf(15.0f, fmaxf(-15.0f, x));
  float e = __expf(2.0f * x);
  return (e - 1.0f) / (e + 1.0f);
}

// Shared GEMM phase: acc += src(64xHH, LDS-staged in 16KB chunks) @ W^T slice.
// wave w: band/mtile mapping is done by caller via colBase/mt.
__device__ __forceinline__ void gemm_phase(
    const _Float16* __restrict__ src, const _Float16* __restrict__ W,
    v4u* shq, int colBase, int tid, int l16, int lh, int mt,
    v8f& acc0, v8f& acc1) {
  const _Float16* sh = (const _Float16*)shq;
  for (int ch = 0; ch < HH / 128; ++ch) {
    __syncthreads();
    {
      int idx = tid;                      // 1024 v4u, 4 per thread
#pragma unroll
      for (int i = 0; i < 4; ++i, idx += 256) {
        int row = idx >> 4, qc = idx & 15;
        shq[idx] = ((const v4u*)src)[row * (HH / 8) + ch * 16 + qc];
      }
    }
    __syncthreads();
#pragma unroll
    for (int kk = 0; kk < 4; ++kk) {
      v16h b  = load_b(W, colBase, HH, ch * 128 + kk * 32, l16, lh);
      v16h a0 = load_a(sh, mt * 16, 128, kk * 32, l16, lh);
      v16h a1 = load_a(sh, (mt + 2) * 16, 128, kk * 32, l16, lh);
      acc0 = wmma16(a0, b, acc0);
      acc1 = wmma16(a1, b, acc1);
    }
  }
}

// ---------------------------------------------------------------------------
// Setup kernels
// ---------------------------------------------------------------------------

// xs = x[:, 1:, :] -> fp16, time-major rows r = t*64 + b, K padded to 320.
__global__ void k_cvt_x(const float* __restrict__ x, _Float16* __restrict__ xf) {
  size_t idx = (size_t)blockIdx.x * blockDim.x + threadIdx.x;
  size_t n = (size_t)ROWS * IPAD;
  if (idx >= n) return;
  int r = (int)(idx / IPAD), k = (int)(idx % IPAD);
  int t = r / BB, b = r % BB;
  float v = 0.0f;
  if (k < II) v = x[((size_t)b * SS + (t + 1)) * II + k];
  xf[idx] = (_Float16)v;
}

// generic fp32 -> fp16 convert with zero padding (dst is dst_rows x dst_cols)
__global__ void k_cvt_w(const float* __restrict__ src, _Float16* __restrict__ dst,
                        int src_rows, int src_cols, int dst_rows, int dst_cols) {
  size_t idx = (size_t)blockIdx.x * blockDim.x + threadIdx.x;
  size_t n = (size_t)dst_rows * dst_cols;
  if (idx >= n) return;
  int r = (int)(idx / dst_cols), c = (int)(idx % dst_cols);
  float v = 0.0f;
  if (r < src_rows && c < src_cols) v = src[(size_t)r * src_cols + c];
  dst[idx] = (_Float16)v;
}

// zero h1/h2 initial buffers, c1/c2, and out[:, 0, :]
__global__ void k_init(_Float16* __restrict__ h1, _Float16* __restrict__ h2,
                       float* __restrict__ c1, float* __restrict__ c2,
                       float* __restrict__ out) {
  int tid = blockIdx.x * blockDim.x + threadIdx.x;   // 65536 threads
  h1[tid] = (_Float16)0.0f;
  h2[tid] = (_Float16)0.0f;
  c1[tid] = 0.0f;
  c2[tid] = 0.0f;
  if (tid < BB * OO) {
    int b = tid / OO, o = tid % OO;
    out[(size_t)b * SS * OO + o] = 0.0f;
  }
}

// ---------------------------------------------------------------------------
// Phase 1: xg1[r, g] = xf16 @ W_ih1f^T + (b_ih1 + b_hh1), stored fp16.
// grid (256 n-tiles, 1023 row-groups of 64), block = 128 (4 waves, 1 mtile each)
// ---------------------------------------------------------------------------
__global__ __launch_bounds__(128) void k_pregemm(
    const _Float16* __restrict__ xf, const _Float16* __restrict__ Wih1f,
    const float* __restrict__ b_ih1, const float* __restrict__ b_hh1,
    _Float16* __restrict__ xg1) {
  int tid = threadIdx.x;
  int lane = tid & 31, l16 = lane & 15, lh = lane >> 4;
  int wave = tid >> 5;
  int nt = blockIdx.x;
  int rowBase = blockIdx.y * 64 + wave * 16;
  v8f acc = {};
#pragma unroll
  for (int kk = 0; kk < IPAD / 32; ++kk) {
    v16h a = load_a(xf, rowBase, IPAD, kk * 32, l16, lh);
    v16h b = load_b(Wih1f, nt * 16, IPAD, kk * 32, l16, lh);
    acc = wmma16(a, b, acc);
  }
  int g = nt * 16 + l16;
  float bias = b_ih1[g] + b_hh1[g];
#pragma unroll
  for (int v = 0; v < 8; ++v) {
    int r = rowBase + v + 8 * lh;
    xg1[(size_t)r * GG + g] = (_Float16)(acc[v] + bias);
  }
}

// ---------------------------------------------------------------------------
// Fused pipeline tick t (one launch per tick, 135 blocks x 256 threads):
//   blocks   0..63  : layer-1 step t      : g = h1(t-1) @ W_hh1^T + xg1[t]
//   blocks  64..127 : layer-2 step t-1    : g = h1(t-1)n @ W_ih2^T
//                                             + h2(t-2) @ W_hh2^T + bias2
//   blocks 128..134 : out-proj step t-2   : out[:, t-1, :] = h2(t-2)n @ W_out^T
// Buffer roles for tick t (double-buffered):
//   h1r = h1b[t&1]     (read by L1 and L2)   h1w = h1b[(t+1)&1] (written by L1)
//   h2r = h2b[(t+1)&1] (read by L2 and OUT)  h2w = h2b[t&1]     (written by L2)
// ---------------------------------------------------------------------------
__global__ __launch_bounds__(256) void k_step(
    const _Float16* __restrict__ Whh1f, const _Float16* __restrict__ Wih2f,
    const _Float16* __restrict__ Whh2f, const _Float16* __restrict__ Woutf,
    const _Float16* __restrict__ xg1,
    const _Float16* __restrict__ h1r, _Float16* __restrict__ h1w,
    const _Float16* __restrict__ h2r, _Float16* __restrict__ h2w,
    float* __restrict__ c1, float* __restrict__ c2,
    const float* __restrict__ b_ih2, const float* __restrict__ b_hh2,
    const float* __restrict__ b_out, float* __restrict__ out, int t) {
  __shared__ v4u  shq[1024];          // 64 x 128 fp16 K-chunk (16 KB)
  __shared__ float sg[4][BB][16];     // gate staging (16 KB)

  int tid = threadIdx.x;
  int lane = tid & 31, l16 = lane & 15, lh = lane >> 4;
  int wave = tid >> 5;
  int bx = blockIdx.x;

  if (bx < 64) {
    // ---------------- layer-1, step t ----------------
    if (t >= TT) return;
    int band = wave & 3, mt = wave >> 2;
    int colBase = band * HH + bx * 16;
    v8f acc0 = {}, acc1 = {};
    gemm_phase(h1r, Whh1f, shq, colBase, tid, l16, lh, mt, acc0, acc1);
#pragma unroll
    for (int v = 0; v < 8; ++v) {
      sg[band][mt * 16 + v + 8 * lh][l16]       = acc0[v];
      sg[band][(mt + 2) * 16 + v + 8 * lh][l16] = acc1[v];
    }
    __syncthreads();
    for (int e = tid; e < BB * 16; e += 256) {
      int m = e >> 4, jl = e & 15;
      int j = bx * 16 + jl;
      const _Float16* xg = xg1 + ((size_t)t * BB + m) * GG;
      float gi = sg[0][m][jl] + (float)xg[j];
      float gf = sg[1][m][jl] + (float)xg[HH + j];
      float gg = sg[2][m][jl] + (float)xg[2 * HH + j];
      float go = sg[3][m][jl] + (float)xg[3 * HH + j];
      float i_ = fsigmoid(gi), f_ = fsigmoid(gf);
      float g_ = ftanh(gg),    o_ = fsigmoid(go);
      size_t sidx = (size_t)m * HH + j;
      float c = f_ * c1[sidx] + i_ * g_;
      c1[sidx] = c;
      h1w[sidx] = (_Float16)(o_ * ftanh(c));
    }
  } else if (bx < 128) {
    // ---------------- layer-2, step t-1 ----------------
    if (t < 1 || t > TT) return;
    int nb = bx - 64;
    int band = wave & 3, mt = wave >> 2;
    int colBase = band * HH + nb * 16;
    v8f acc0 = {}, acc1 = {};
    gemm_phase(h1r, Wih2f, shq, colBase, tid, l16, lh, mt, acc0, acc1);
    gemm_phase(h2r, Whh2f, shq, colBase, tid, l16, lh, mt, acc0, acc1);
#pragma unroll
    for (int v = 0; v < 8; ++v) {
      sg[band][mt * 16 + v + 8 * lh][l16]       = acc0[v];
      sg[band][(mt + 2) * 16 + v + 8 * lh][l16] = acc1[v];
    }
    __syncthreads();
    for (int e = tid; e < BB * 16; e += 256) {
      int m = e >> 4, jl = e & 15;
      int j = nb * 16 + jl;
      float gi = sg[0][m][jl] + b_ih2[j]          + b_hh2[j];
      float gf = sg[1][m][jl] + b_ih2[HH + j]     + b_hh2[HH + j];
      float gg = sg[2][m][jl] + b_ih2[2 * HH + j] + b_hh2[2 * HH + j];
      float go = sg[3][m][jl] + b_ih2[3 * HH + j] + b_hh2[3 * HH + j];
      float i_ = fsigmoid(gi), f_ = fsigmoid(gf);
      float g_ = ftanh(gg),    o_ = fsigmoid(go);
      size_t sidx = (size_t)m * HH + j;
      float c = f_ * c2[sidx] + i_ * g_;
      c2[sidx] = c;
      h2w[sidx] = (_Float16)(o_ * ftanh(c));
    }
  } else {
    // ---------------- out projection, step t-2 ----------------
    if (t < 2) return;
    int nt = bx - 128;
    if (wave >= 4) return;            // 4 mtiles; no barriers on this path
    v8f acc = {};
#pragma unroll 4
    for (int kk = 0; kk < HH / 32; ++kk) {
      v16h a = load_a(h2r, wave * 16, HH, kk * 32, l16, lh);
      v16h b = load_b(Woutf, nt * 16, HH, kk * 32, l16, lh);
      acc = wmma16(a, b, acc);
    }
    int n = nt * 16 + l16;
#pragma unroll
    for (int v = 0; v < 8; ++v) {
      int m = wave * 16 + v + 8 * lh;
      if (n < OO)
        out[(size_t)m * SS * OO + (size_t)(t - 1) * OO + n] = acc[v] + b_out[n];
    }
  }
}

// ---------------------------------------------------------------------------
// Host orchestration
// ---------------------------------------------------------------------------
extern "C" void kernel_launch(void* const* d_in, const int* in_sizes, int n_in,
                              void* d_out, int out_size, void* d_ws, size_t ws_size,
                              hipStream_t stream) {
  (void)in_sizes; (void)n_in; (void)out_size; (void)ws_size;
  const float* x      = (const float*)d_in[0];
  const float* W_ih1  = (const float*)d_in[1];
  const float* W_hh1  = (const float*)d_in[2];
  const float* b_ih1  = (const float*)d_in[3];
  const float* b_hh1  = (const float*)d_in[4];
  const float* W_ih2  = (const float*)d_in[5];
  const float* W_hh2  = (const float*)d_in[6];
  const float* b_ih2  = (const float*)d_in[7];
  const float* b_hh2  = (const float*)d_in[8];
  const float* W_out  = (const float*)d_in[9];
  const float* b_out  = (const float*)d_in[10];
  float* out = (float*)d_out;

  char* ws = (char*)d_ws;
  size_t off = 0;
  auto alloc = [&](size_t bytes) -> void* {
    void* p = ws + off;
    off += (bytes + 255) & ~(size_t)255;
    return p;
  };
  _Float16* xf16   = (_Float16*)alloc((size_t)ROWS * IPAD * 2);
  _Float16* Wih1f  = (_Float16*)alloc((size_t)GG * IPAD * 2);
  _Float16* Whh1f  = (_Float16*)alloc((size_t)GG * HH * 2);
  _Float16* Wih2f  = (_Float16*)alloc((size_t)GG * HH * 2);
  _Float16* Whh2f  = (_Float16*)alloc((size_t)GG * HH * 2);
  _Float16* Woutf  = (_Float16*)alloc((size_t)OPAD * HH * 2);
  _Float16* xg1    = (_Float16*)alloc((size_t)ROWS * GG * 2);
  _Float16* h1b[2] = { (_Float16*)alloc(BB * HH * 2), (_Float16*)alloc(BB * HH * 2) };
  _Float16* h2b[2] = { (_Float16*)alloc(BB * HH * 2), (_Float16*)alloc(BB * HH * 2) };
  float* c1 = (float*)alloc(BB * HH * 4);
  float* c2 = (float*)alloc(BB * HH * 4);

  // --- setup ---
  {
    size_t n = (size_t)ROWS * IPAD;
    k_cvt_x<<<dim3((unsigned)((n + 255) / 256)), 256, 0, stream>>>(x, xf16);
  }
  auto cvt = [&](const float* s, _Float16* d, int sr, int sc, int dr, int dc) {
    size_t n = (size_t)dr * dc;
    k_cvt_w<<<dim3((unsigned)((n + 255) / 256)), 256, 0, stream>>>(s, d, sr, sc, dr, dc);
  };
  cvt(W_ih1, Wih1f, GG, II, GG, IPAD);
  cvt(W_hh1, Whh1f, GG, HH, GG, HH);
  cvt(W_ih2, Wih2f, GG, HH, GG, HH);
  cvt(W_hh2, Whh2f, GG, HH, GG, HH);
  cvt(W_out, Woutf, OO, HH, OPAD, HH);
  k_init<<<dim3(256), 256, 0, stream>>>(h1b[0], h2b[0], c1, c2, out);

  // --- phase 1: input-gate precompute GEMM ---
  k_pregemm<<<dim3(GG / 16, TT), 128, 0, stream>>>(xf16, Wih1f, b_ih1, b_hh1, xg1);

  // --- phase 2: software-pipelined scan, one launch per tick ---
  for (int t = 0; t <= TT + 1; ++t) {
    const _Float16* h1r = h1b[t & 1];
    _Float16*       h1w = h1b[(t + 1) & 1];
    const _Float16* h2r = h2b[(t + 1) & 1];
    _Float16*       h2w = h2b[t & 1];
    k_step<<<dim3(64 + 64 + OPAD / 16), 256, 0, stream>>>(
        Whh1f, Wih2f, Whh2f, Woutf, xg1, h1r, h1w, h2r, h2w,
        c1, c2, b_ih2, b_hh2, b_out, out, t);
  }
}
